// WLinear_74199855006131
// MI455X (gfx1250) — matrix operand order repaired and checked by reading, not compiled, not run on hardware
//
#include <hip/hip_runtime.h>

typedef __attribute__((ext_vector_type(16))) _Float16 v16h;
typedef __attribute__((ext_vector_type(8)))  _Float16 v8h;
typedef __attribute__((ext_vector_type(2)))  _Float16 v2h;
typedef __attribute__((ext_vector_type(8)))  float    v8f;

namespace {
constexpr int M_TOT = 4096;
constexpr int N_TOT = 11008;
constexpr int K_TOT = 4096;
constexpr int QK    = K_TOT / 2;    // int32 per qweight row (2 nibbles used per int32)
constexpr int NGRP  = K_TOT / 128;  // 32 quant groups per output row
constexpr int BM = 128;
constexpr int BN = 128;
constexpr int BK = 32;
constexpr int NT = K_TOT / BK;      // 128 k-tiles, 4 per quant group
constexpr int LDT = 40;             // padded halves per LDS tile row (80B, 16B-aligned rows)
}

// Dequant 1 packed int32 -> 2 f16 weights via magic-bias + packed FMA:
//   t = {0x6400|hi , 0x6400|lo} = {1024+hi, 1024+lo} as f16x2
//   r = t*s + (-(1024+z)*s)  ->  (q - z)*s   (one v_pk_fma_f16)
__device__ __forceinline__ unsigned dq2(unsigned q, v2h sv, v2h ov) {
    unsigned t = (q & 0xFu) | ((q << 12) & 0x000F0000u) | 0x64006400u;
    v2h v = __builtin_bit_cast(v2h, t);
    v2h r = v * sv + ov;
    return __builtin_bit_cast(unsigned, r);
}

__global__ __launch_bounds__(256)
void wlinear_w4a16_wmma(const _Float16* __restrict__ x,
                        const int*      __restrict__ qw,
                        const _Float16* __restrict__ wscale,
                        const int*      __restrict__ wzero,
                        const _Float16* __restrict__ bias,
                        float*          __restrict__ out)
{
    __shared__ __align__(16) _Float16 As[2][BM * LDT];   // activations [m][k], double buffered
    __shared__ __align__(16) _Float16 Ws[2][BN * LDT];   // dequantized W [n][k], double buffered

    const int tid  = threadIdx.x;
    const int lane = tid & 31;          // wave32
    const int wave = tid >> 5;          // 0..7
    const int wy   = wave >> 1;         // 0..3 -> M
    const int wx   = wave & 1;          // 0..1 -> N
    const int mbase = wy * 32;
    const int nbase = wx * 64;
    const int lrow  = lane & 15;
    const int hi16  = lane >> 4;        // 0 or 1

    const int bm = blockIdx.y * BM;
    const int bn = blockIdx.x * BN;

    // ---- tile loader mapping: thread -> (row, 16-half segment) ----
    const int lr  = tid >> 1;           // 0..127
    const int seg = tid & 1;            // 0..1

    const int*      qrow = qw + (size_t)(bn + lr) * QK + seg * 8;
    const _Float16* srow = wscale + (size_t)(bn + lr) * NGRP;
    const int*      zrow = wzero  + (size_t)(bn + lr) * NGRP;

    // byte offset of this thread's x chunk within the matrix (fits 32 bits: 33.5MB)
    const unsigned xoff_row = ((unsigned)(bm + lr) * (unsigned)K_TOT + (unsigned)(seg * 16)) * 2u;

    // ---- async copy of the A (x) tile: global -> LDS, ASYNCcnt-tracked ----
    auto asyncA = [&](int k0, int buf) {
        unsigned lds  = (unsigned)(uintptr_t)(&As[buf][lr * LDT + seg * 16]);
        unsigned voff = xoff_row + (unsigned)k0 * 2u;
        asm volatile(
            "global_load_async_to_lds_b128 %0, %1, %2 offset:0\n\t"
            "global_load_async_to_lds_b128 %0, %1, %2 offset:16"
            :: "v"(lds), "v"(voff), "s"(x) : "memory");
    };

    // ---- staged qweight (register pipeline, one tile ahead) ----
    int4 q0, q1;
    auto stageQ = [&](int k0) {
        const int4* qp = (const int4*)(qrow + (k0 >> 1));
        q0 = qp[0];
        q1 = qp[1];
    };

    auto writeB = [&](int buf, v2h sv, v2h ov) {
        uint4 s0, s1;
        s0.x = dq2((unsigned)q0.x, sv, ov);
        s0.y = dq2((unsigned)q0.y, sv, ov);
        s0.z = dq2((unsigned)q0.z, sv, ov);
        s0.w = dq2((unsigned)q0.w, sv, ov);
        s1.x = dq2((unsigned)q1.x, sv, ov);
        s1.y = dq2((unsigned)q1.y, sv, ov);
        s1.z = dq2((unsigned)q1.z, sv, ov);
        s1.w = dq2((unsigned)q1.w, sv, ov);
        uint4* wsl = (uint4*)(&Ws[buf][lr * LDT + seg * 16]);
        wsl[0] = s0;
        wsl[1] = s1;
    };

    // fragment loaders (LDS -> VGPR), per ISA 16-bit A/B layouts
    auto ldfragA = [&](const _Float16* p) -> v16h {
        v8h lo = *(const v8h*)(p);          // K = kba .. kba+7
        v8h hi = *(const v8h*)(p + 16);     // K = kba+16 .. kba+23
        v16h r;
#pragma unroll
        for (int i = 0; i < 8; ++i) { r[i] = lo[i]; r[i + 8] = hi[i]; }
        return r;
    };
    auto ldfragB = [&](const _Float16* p) -> v16h {
        v8h lo = *(const v8h*)(p);          // K = kbb .. kbb+7
        v8h hi = *(const v8h*)(p + 16);     // K = kbb+16 .. kbb+23
        v16h r;
#pragma unroll
        for (int i = 0; i < 8; ++i) { r[i] = lo[i]; r[i + 8] = hi[i]; }
        return r;
    };

    v8f acc[2][4];
    {
        v8f z{};
#pragma unroll
        for (int i = 0; i < 2; ++i)
#pragma unroll
            for (int j = 0; j < 4; ++j) acc[i][j] = z;
    }

    const int kba = hi16 * 8;    // A: lanes 16-31 hold K 8..15 / 24..31
    const int kbb = hi16 * 8;    // B: lanes 16-31 hold K 8..15 / 24..31

    auto compute = [&](int buf) {
        v16h afrag0 = ldfragA(&As[buf][(mbase + lrow) * LDT + kba]);
        v16h afrag1 = ldfragA(&As[buf][(mbase + 16 + lrow) * LDT + kba]);
        v16h bfrag[4];
#pragma unroll
        for (int j = 0; j < 4; ++j)
            bfrag[j] = ldfragB(&Ws[buf][(nbase + 16 * j + lrow) * LDT + kbb]);
#pragma unroll
        for (int j = 0; j < 4; ++j) {
            acc[0][j] = __builtin_amdgcn_wmma_f32_16x16x32_f16(
                false, afrag0, false, bfrag[j], (short)0, acc[0][j], false, false);
            acc[1][j] = __builtin_amdgcn_wmma_f32_16x16x32_f16(
                false, afrag1, false, bfrag[j], (short)0, acc[1][j], false, false);
        }
    };

    // ---- prologue: start tile 0, stage group-0 quant params ----
    asyncA(0, 0);
    stageQ(0);
    _Float16 sc_cur = srow[0];
    int      zp_cur = zrow[0];
    _Float16 sc_nxt = sc_cur;
    int      zp_nxt = zp_cur;

    for (int g = 0; g < NGRP; ++g) {
        // group constants, built once per 4 k-tiles
        v2h sv; sv[0] = sc_cur; sv[1] = sc_cur;
        const _Float16 zb = (_Float16)(zp_cur + 1024);
        const _Float16 oh = (_Float16)0.0f - zb * sc_cur;
        v2h ov; ov[0] = oh; ov[1] = oh;

        // prefetch next group's scalars (4 tiles of compute to hide them)
        if (g + 1 < NGRP) {
            sc_nxt = srow[g + 1];
            zp_nxt = zrow[g + 1];
        }

#pragma unroll
        for (int i = 0; i < 4; ++i) {
            const int t   = g * 4 + i;
            const int k0  = t * BK;
            const int buf = i & 1;                        // compile-time after unroll

            writeB(buf, sv, ov);                          // dequant -> LDS (DScnt)
            asm volatile("s_wait_asynccnt 0x0" ::: "memory"); // our A-tile copy landed
            __syncthreads();                              // tile visible to all waves

            if (t + 1 < NT) {                             // start tile t+1 during compute
                asyncA(k0 + BK, buf ^ 1);
                stageQ(k0 + BK);
            }
            compute(buf);
        }

        sc_cur = sc_nxt;
        zp_cur = zp_nxt;
    }

    // ---- epilogue: acc(f32) + bias, store f32 ----
    // C/D layout: VGPR r -> M = r (lanes 0-15) / r+8 (lanes 16-31); N = lane&15
    const int rowsel = hi16 * 8;
#pragma unroll
    for (int j = 0; j < 4; ++j) {
        const int gcol = bn + nbase + 16 * j + lrow;
        const float bf = (float)bias[gcol];
#pragma unroll
        for (int i = 0; i < 2; ++i) {
            const int grow0 = bm + mbase + i * 16 + rowsel;
#pragma unroll
            for (int r = 0; r < 8; ++r) {
                out[(size_t)(grow0 + r) * N_TOT + gcol] = acc[i][j][r] + bf;
            }
        }
    }
}

extern "C" void kernel_launch(void* const* d_in, const int* in_sizes, int n_in,
                              void* d_out, int out_size, void* d_ws, size_t ws_size,
                              hipStream_t stream) {
    (void)in_sizes; (void)n_in; (void)out_size; (void)d_ws; (void)ws_size;
    const _Float16* x      = (const _Float16*)d_in[0];   // fp16 [4096, 4096]
    const int*      qw     = (const int*)d_in[1];        // int32 [11008, 2048]
    const _Float16* wscale = (const _Float16*)d_in[2];   // fp16 [11008, 32]
    const int*      wzero  = (const int*)d_in[3];        // int32 [11008, 32]
    const _Float16* bias   = (const _Float16*)d_in[4];   // fp16 [11008]
    float*          out    = (float*)d_out;              // [4096, 11008]

    dim3 grid(N_TOT / BN, M_TOT / BM);   // (86, 32)
    dim3 block(256);
    wlinear_w4a16_wmma<<<grid, block, 0, stream>>>(x, qw, wscale, wzero, bias, out);
}
